// GroupedQueryAttention_55576876810886
// MI455X (gfx1250) — matrix operand order repaired and checked by reading, not compiled, not run on hardware
//
#include <hip/hip_runtime.h>
#include <hip/hip_bf16.h>

typedef _Float16 half_t;
typedef __attribute__((ext_vector_type(16))) _Float16 v16h;
typedef __attribute__((ext_vector_type(8)))  _Float16 v8h;
typedef __attribute__((ext_vector_type(8)))  float    v8f;

#define WMMA16x16x32(a, b, c) \
  __builtin_amdgcn_wmma_f32_16x16x32_f16(false, (a), false, (b), (short)0, (c), false, false)

// ---- constants ----
#define BB 4
#define TT 2048
#define CC 1024
#define NH 16
#define KVH 4
#define HD 64
#define MM (BB * TT)          // 8192
#define NQKV (CC + 2 * KVH * HD)  // 1536

// A-matrix 16x32 f16 fragment: lane m = l&15, group g = l>>4.
// halves 0..7  -> K = 16g .. 16g+7
// halves 8..15 -> K = 16g+8 .. 16g+15
__device__ inline v16h load_a_frag(const half_t* base, int row_stride, int lane) {
  int m = lane & 15, g = lane >> 4;
  const half_t* p = base + (size_t)m * row_stride;
  v8h lo = *(const v8h*)(p + 16 * g);
  v8h hi = *(const v8h*)(p + 16 * g + 8);
  return __builtin_shufflevector(lo, hi, 0,1,2,3,4,5,6,7,8,9,10,11,12,13,14,15);
}

// B-operand (32x16, K x N) fragment: lane holds column n = l&15,
// halves = contiguous K range [16g, 16g+15]. Source stored N-major
// (row n contiguous in K), i.e. exactly a torch [out,in] weight row.
__device__ inline v16h load_b_frag(const half_t* base, int row_stride, int lane) {
  int n = lane & 15, g = lane >> 4;
  return *(const v16h*)(base + (size_t)n * row_stride + 16 * g);
}

// ---------------- fp32 -> f16 conversion ----------------
__global__ __launch_bounds__(256) void cvt_f32_f16(const float* __restrict__ src,
                                                   half_t* __restrict__ dst, int n) {
  int i = blockIdx.x * blockDim.x + threadIdx.x;
  int stride = gridDim.x * blockDim.x;
  for (; i < n; i += stride) dst[i] = (half_t)src[i];
}

// ---------------- fused QKV projection ----------------
// y = x @ W^T for W in {Wq, Wk, Wv} (N = 1024 + 256 + 256).
// Block: 256 threads = 8 waves; block tile 128(M) x 64(N); wave tile 32x32.
__global__ __launch_bounds__(256) void qkv_gemm(
    const half_t* __restrict__ x,   // [8192][1024]
    const half_t* __restrict__ wq,  // [1024][1024]
    const half_t* __restrict__ wk,  // [256][1024]
    const half_t* __restrict__ wv,  // [256][1024]
    half_t* __restrict__ qh,        // [B][NH][T][64]
    half_t* __restrict__ kh,        // [B][KVH][T][64]
    half_t* __restrict__ vt)        // [B][KVH][64][T]  (transposed V)
{
  const int lane = threadIdx.x & 31;
  const int wave = threadIdx.x >> 5;
  const int tileM = blockIdx.x * 128 + (wave >> 1) * 32;
  const int tileN = blockIdx.y * 64 + (wave & 1) * 32;

  v8f acc[2][2] = {{}, {}};
  for (int k0 = 0; k0 < CC; k0 += 32) {
    v16h a0 = load_a_frag(x + (size_t)tileM * CC + k0, CC, lane);
    v16h a1 = load_a_frag(x + (size_t)(tileM + 16) * CC + k0, CC, lane);
#pragma unroll
    for (int j = 0; j < 2; ++j) {
      int n0 = tileN + j * 16;
      const half_t* wbase; int nl;
      if (n0 < CC)            { wbase = wq; nl = n0; }
      else if (n0 < CC + 256) { wbase = wk; nl = n0 - CC; }
      else                    { wbase = wv; nl = n0 - CC - 256; }
      v16h b = load_b_frag(wbase + (size_t)nl * CC + k0, CC, lane);
      acc[0][j] = WMMA16x16x32(a0, b, acc[0][j]);
      acc[1][j] = WMMA16x16x32(a1, b, acc[1][j]);
    }
  }

  // C-layout: lane n = l&15 ; vgpr r -> row r + 8*(l>>4)
  const int g = lane >> 4;
#pragma unroll
  for (int i = 0; i < 2; ++i) {
#pragma unroll
    for (int j = 0; j < 2; ++j) {
      int n = tileN + j * 16 + (lane & 15);
#pragma unroll
      for (int r = 0; r < 8; ++r) {
        int gm = tileM + i * 16 + r + 8 * g;
        int b_ = gm >> 11;            // /2048
        int t  = gm & (TT - 1);
        half_t val = (half_t)acc[i][j][r];
        if (n < CC) {                 // Q
          int h = n >> 6, d = n & 63;
          qh[(((size_t)(b_ * NH + h)) * TT + t) * HD + d] = val;
        } else if (n < CC + 256) {    // K
          int c = n - CC, kv = c >> 6, d = c & 63;
          kh[(((size_t)(b_ * KVH + kv)) * TT + t) * HD + d] = val;
        } else {                      // V, stored transposed [d][t]
          int c = n - CC - 256, kv = c >> 6, d = c & 63;
          vt[(((size_t)(b_ * KVH + kv)) * HD + d) * TT + t] = val;
        }
      }
    }
  }
}

// ---------------- causal flash attention ----------------
// grid (T/128, NH, B); 8 waves/block, each wave = 16 query rows.
__global__ __launch_bounds__(256) void attn(
    const half_t* __restrict__ qh, const half_t* __restrict__ kh,
    const half_t* __restrict__ vt, half_t* __restrict__ ao /* [B][T][C] f16 */)
{
  __shared__ __align__(32) half_t plds[8][16 * 32];
  const int lane = threadIdx.x & 31;
  const int wave = threadIdx.x >> 5;
  const int qb = blockIdx.x * 128 + wave * 16;
  const int h = blockIdx.y, b = blockIdx.z;
  const int kv = h >> 2;  // GROUPS = 4

  const half_t* Q = qh + (((size_t)(b * NH + h)) * TT + qb) * HD;
  const half_t* K = kh + ((size_t)(b * KVH + kv)) * TT * HD;   // [t][64]
  const half_t* V = vt + ((size_t)(b * KVH + kv)) * HD * TT;   // [d][T]
  half_t* pl = plds[wave];

  // Q fragments live in registers for the whole key loop
  v16h qa0 = load_a_frag(Q + 0, HD, lane);
  v16h qa1 = load_a_frag(Q + 32, HD, lane);

  v8f o[4] = {};
  float mrow[8], lrow[8];
#pragma unroll
  for (int r = 0; r < 8; ++r) { mrow[r] = -__builtin_inff(); lrow[r] = 0.0f; }

  const float scale = 0.125f;  // 1/sqrt(64)
  const int g = lane >> 4, nn = lane & 15;
  const int nkb = (qb + 15) / 32 + 1;  // causal: only blocks touching <= diagonal

  for (int kb = 0; kb < nkb; ++kb) {
    const int key0 = kb * 32;
    // S = Q K^T  (two 16-key tiles, K-dim = head_dim 64 in two steps)
    v8f s[2] = {};
#pragma unroll
    for (int nt = 0; nt < 2; ++nt) {
      v16h b0 = load_b_frag(K + (size_t)(key0 + nt * 16) * HD + 0, HD, lane);
      v16h b1 = load_b_frag(K + (size_t)(key0 + nt * 16) * HD + 32, HD, lane);
      s[nt] = WMMA16x16x32(qa0, b0, s[nt]);
      s[nt] = WMMA16x16x32(qa1, b1, s[nt]);
    }
    // scale + causal mask (element: key = key0+nt*16+nn, row = qb+r+8g)
#pragma unroll
    for (int nt = 0; nt < 2; ++nt)
#pragma unroll
      for (int r = 0; r < 8; ++r) {
        int key = key0 + nt * 16 + nn;
        int qr = qb + r + 8 * g;
        float sv = s[nt][r] * scale;
        s[nt][r] = (key > qr) ? -__builtin_inff() : sv;
      }
    // online softmax per row (rows live across 16-lane halves)
    float nm[8], alpha[8], tsum[8];
#pragma unroll
    for (int r = 0; r < 8; ++r) {
      float tm = fmaxf(s[0][r], s[1][r]);
#pragma unroll
      for (int mk = 8; mk >= 1; mk >>= 1) tm = fmaxf(tm, __shfl_xor(tm, mk, 32));
      float newm = fmaxf(mrow[r], tm);
      nm[r] = newm;
      alpha[r] = __expf(mrow[r] - newm);
      float p0 = __expf(s[0][r] - newm);
      float p1 = __expf(s[1][r] - newm);
      s[0][r] = p0; s[1][r] = p1;
      float ts = p0 + p1;
#pragma unroll
      for (int mk = 8; mk >= 1; mk >>= 1) ts += __shfl_xor(ts, mk, 32);
      tsum[r] = ts;
    }
#pragma unroll
    for (int r = 0; r < 8; ++r) {
      lrow[r] = lrow[r] * alpha[r] + tsum[r];
      mrow[r] = nm[r];
    }
#pragma unroll
    for (int dt = 0; dt < 4; ++dt)
#pragma unroll
      for (int r = 0; r < 8; ++r) o[dt][r] *= alpha[r];

    // transpose P (C-layout) -> A-fragment via per-wave LDS (wave-local, in-order DS)
#pragma unroll
    for (int nt = 0; nt < 2; ++nt)
#pragma unroll
      for (int r = 0; r < 8; ++r)
        pl[(r + 8 * g) * 32 + nt * 16 + nn] = (half_t)s[nt][r];
    v16h pa = load_a_frag(pl, 32, lane);

    // O += P @ V  (B operand from transposed V: lane = d, halves = keys)
#pragma unroll
    for (int dt = 0; dt < 4; ++dt) {
      v16h vb = load_b_frag(V + (size_t)(dt * 16) * TT + key0, TT, lane);
      o[dt] = WMMA16x16x32(pa, vb, o[dt]);
    }
  }

  // epilogue: normalize, write into [B][T][C] (head-concat) as f16
#pragma unroll
  for (int dt = 0; dt < 4; ++dt)
#pragma unroll
    for (int r = 0; r < 8; ++r) {
      int t = qb + r + 8 * g;
      int col = h * HD + dt * 16 + nn;
      ao[((size_t)b * TT + t) * CC + col] = (half_t)(o[dt][r] / lrow[r]);
    }
}

// ---------------- output projection ----------------
__global__ __launch_bounds__(256) void out_gemm(
    const half_t* __restrict__ ao,   // [8192][1024]
    const half_t* __restrict__ wo,   // [1024][1024]
    float* __restrict__ out)         // [8192][1024] fp32
{
  const int lane = threadIdx.x & 31;
  const int wave = threadIdx.x >> 5;
  const int tileM = blockIdx.x * 128 + (wave >> 1) * 32;
  const int tileN = blockIdx.y * 64 + (wave & 1) * 32;

  v8f acc[2][2] = {{}, {}};
  for (int k0 = 0; k0 < CC; k0 += 32) {
    v16h a0 = load_a_frag(ao + (size_t)tileM * CC + k0, CC, lane);
    v16h a1 = load_a_frag(ao + (size_t)(tileM + 16) * CC + k0, CC, lane);
#pragma unroll
    for (int j = 0; j < 2; ++j) {
      v16h b = load_b_frag(wo + (size_t)(tileN + j * 16) * CC + k0, CC, lane);
      acc[0][j] = WMMA16x16x32(a0, b, acc[0][j]);
      acc[1][j] = WMMA16x16x32(a1, b, acc[1][j]);
    }
  }
  const int g = lane >> 4;
#pragma unroll
  for (int i = 0; i < 2; ++i)
#pragma unroll
    for (int j = 0; j < 2; ++j) {
      int n = tileN + j * 16 + (lane & 15);
#pragma unroll
      for (int r = 0; r < 8; ++r) {
        int gm = tileM + i * 16 + r + 8 * g;
        out[(size_t)gm * CC + n] = acc[i][j][r];
      }
    }
}

extern "C" void kernel_launch(void* const* d_in, const int* in_sizes, int n_in,
                              void* d_out, int out_size, void* d_ws, size_t ws_size,
                              hipStream_t stream) {
  const float* x  = (const float*)d_in[0];
  const float* Wq = (const float*)d_in[1];
  const float* Wk = (const float*)d_in[2];
  const float* Wv = (const float*)d_in[3];
  const float* Wo = (const float*)d_in[4];
  // d_in[5] = mask: causal mask applied analytically in-kernel.
  float* out = (float*)d_out;

  // workspace carve-up (256B aligned)
  size_t off = 0;
  auto carve = [&](size_t elems) {
    size_t o = off;
    off += (elems * sizeof(half_t) + 255) & ~(size_t)255;
    return (half_t*)((char*)d_ws + o);
  };
  half_t* xh  = carve((size_t)MM * CC);
  half_t* wqh = carve((size_t)CC * CC);
  half_t* wkh = carve((size_t)KVH * HD * CC);
  half_t* wvh = carve((size_t)KVH * HD * CC);
  half_t* woh = carve((size_t)CC * CC);
  half_t* qh  = carve((size_t)BB * NH * TT * HD);
  half_t* kh  = carve((size_t)BB * KVH * TT * HD);
  half_t* vt  = carve((size_t)BB * KVH * HD * TT);
  half_t* ao  = carve((size_t)MM * CC);

  // stage 0: precision convert once (everything then lives in the 192MB L2)
  cvt_f32_f16<<<2048, 256, 0, stream>>>(x,  xh,  MM * CC);
  cvt_f32_f16<<<1024, 256, 0, stream>>>(Wq, wqh, CC * CC);
  cvt_f32_f16<<<512,  256, 0, stream>>>(Wk, wkh, KVH * HD * CC);
  cvt_f32_f16<<<512,  256, 0, stream>>>(Wv, wvh, KVH * HD * CC);
  cvt_f32_f16<<<1024, 256, 0, stream>>>(Wo, woh, CC * CC);

  // stage 1: fused QKV projection (N = 1536)
  qkv_gemm<<<dim3(MM / 128, NQKV / 64), 256, 0, stream>>>(xh, wqh, wkh, wvh, qh, kh, vt);

  // stage 2: causal GQA flash attention
  attn<<<dim3(TT / 128, NH, BB), 256, 0, stream>>>(qh, kh, vt, ao);

  // stage 3: output projection -> fp32 result
  out_gemm<<<dim3(MM / 128, CC / 64), 256, 0, stream>>>(ao, woh, out);
}
